// PerformerAttention_45595372814569
// MI455X (gfx1250) — matrix operand order repaired and checked by reading, not compiled, not run on hardware
//
#include <hip/hip_runtime.h>

#define DEVI __device__ __forceinline__

typedef __bf16 bf16;
typedef __bf16 v16bf __attribute__((ext_vector_type(16)));
typedef __bf16 v8bf  __attribute__((ext_vector_type(8)));
typedef float  v8f   __attribute__((ext_vector_type(8)));

namespace cfg {
constexpr int B = 4, N = 4096, D = 512, H = 8, DH = 64, M = 512;
constexpr int BN = B * N;   // 16384
constexpr int BH = B * H;   // 32
constexpr float NORM  = 0.35355339059327373f;   // DH^-0.25
constexpr float RATIO = 0.04419417382415922f;   // M^-0.5
constexpr float KEPS  = 1e-4f;
constexpr float LNEPS = 1e-12f;
}

#if defined(__gfx1250__)
#define ASYNC_LDS 1
#else
#define ASYNC_LDS 0
#endif

// ---------------- WMMA fragment helpers (16x16x32 bf16, wave32) -------------
// A operand: 16(M)x32(K). lane l (l&15 = row, l>>4 = hi) holds K-chunks
//   [hi*8 .. hi*8+7] and [16+hi*8 .. 16+hi*8+7]  (two 16B loads)
DEVI v16bf load_a_frag(const bf16* base, int ld, int row, int k0, int lrow, int lhi) {
  const bf16* p = base + (size_t)(row + lrow) * ld + k0 + lhi * 8;
  v8bf lo = *reinterpret_cast<const v8bf*>(p);
  v8bf hi = *reinterpret_cast<const v8bf*>(p + 16);
  v16bf a;
#pragma unroll
  for (int i = 0; i < 8; ++i) { a[i] = lo[i]; a[i + 8] = hi[i]; }
  return a;
}
// B operand: 32(K)x16(N). lane l holds column (l&15), K-chunks
//   [hi*8 .. hi*8+7] and [16+hi*8 .. 16+hi*8+7]; source rows are the B-columns.
DEVI v16bf load_b_frag(const bf16* base, int ld, int col, int k0, int lcol, int lhi) {
  const bf16* p = base + (size_t)(col + lcol) * ld + k0 + lhi * 8;
  v8bf lo = *reinterpret_cast<const v8bf*>(p);
  v8bf hi = *reinterpret_cast<const v8bf*>(p + 16);
  v16bf b;
#pragma unroll
  for (int i = 0; i < 8; ++i) { b[i] = lo[i]; b[i + 8] = hi[i]; }
  return b;
}
DEVI v8f wmma_bf16(v16bf a, v16bf b, v8f c) {
  return __builtin_amdgcn_wmma_f32_16x16x32_bf16(false, a, false, b, (short)0, c,
                                                 false, false);
}
DEVI v8f vzero8() {
  v8f z;
#pragma unroll
  for (int i = 0; i < 8; ++i) z[i] = 0.f;
  return z;
}

DEVI float wave_max(float v) {
#pragma unroll
  for (int m = 1; m < 32; m <<= 1) v = fmaxf(v, __shfl_xor(v, m, 32));
  return v;
}
DEVI float wave_sum(float v) {
#pragma unroll
  for (int m = 1; m < 32; m <<= 1) v += __shfl_xor(v, m, 32);
  return v;
}

#if ASYNC_LDS
// async global -> LDS copy, 16 bytes per lane, tracked by ASYNCcnt
DEVI void async_b128(unsigned ldsoff, const void* g) {
  asm volatile("global_load_async_to_lds_b128 %0, %1, off"
               :: "v"(ldsoff), "v"(g) : "memory");
}
DEVI void wait_async0() { asm volatile("s_wait_asynccnt 0x0" ::: "memory"); }
DEVI unsigned lds_off(const void* p) { return (unsigned)(size_t)p; }

// stage one 128x32 A tile + 128x32 B tile (bf16) into LDS (row-major, ld=32)
DEVI void issue_tile(const bf16* __restrict__ A, const bf16* __restrict__ Bw,
                     int ldg, bf16* sA, bf16* sB,
                     int rowBlock, int colBlock, int k0, int tid) {
#pragma unroll
  for (int h = 0; h < 2; ++h) {
    int c = tid + h * 256;          // 512 chunks of 16B per tile
    int row = c >> 2, seg = (c & 3) * 8;
    async_b128(lds_off(sA + row * 32 + seg),
               A + (size_t)(rowBlock + row) * ldg + k0 + seg);
    async_b128(lds_off(sB + row * 32 + seg),
               Bw + (size_t)(colBlock + row) * ldg + k0 + seg);
  }
}
#endif

// ---------------- elementwise converters ------------------------------------
__global__ void k_cvt_bf16(const float* __restrict__ s, bf16* __restrict__ d, int n) {
  int i = blockIdx.x * 256 + threadIdx.x;
  if (i < n) d[i] = (bf16)s[i];
}

// ---------------- QKV projection: Y = X @ W^T, scatter to [B,H,N,DH] --------
__global__ __launch_bounds__(256) void k_gemm_qkv(const bf16* __restrict__ X,
                                                  const bf16* __restrict__ W,
                                                  const float* __restrict__ bias,
                                                  bf16* __restrict__ out, float scale) {
  using namespace cfg;
  __shared__ bf16 sA[2 * 128 * 32];
  __shared__ bf16 sB[2 * 128 * 32];
  const int tid = threadIdx.x, wave = tid >> 5, lane = tid & 31;
  const int lcol = lane & 15, lhi = lane >> 4;
  const int wm = wave & 3, wn = wave >> 2;
  const int rowBlock = blockIdx.x * 128, colBlock = blockIdx.y * 128;
  const int rowBase = rowBlock + wm * 32, colBase = colBlock + wn * 64;
  v8f c[2][4];
#pragma unroll
  for (int i = 0; i < 2; ++i)
#pragma unroll
    for (int j = 0; j < 4; ++j) c[i][j] = vzero8();

#if ASYNC_LDS
  issue_tile(X, W, D, sA, sB, rowBlock, colBlock, 0, tid);
  for (int ks = 0; ks < D / 32; ++ks) {
    wait_async0();
    __syncthreads();
    bf16* curA = sA + (ks & 1) * (128 * 32);
    bf16* curB = sB + (ks & 1) * (128 * 32);
    if (ks + 1 < D / 32)
      issue_tile(X, W, D, sA + ((ks + 1) & 1) * (128 * 32),
                 sB + ((ks + 1) & 1) * (128 * 32),
                 rowBlock, colBlock, (ks + 1) * 32, tid);
    v16bf a0 = load_a_frag(curA, 32, wm * 32,      0, lcol, lhi);
    v16bf a1 = load_a_frag(curA, 32, wm * 32 + 16, 0, lcol, lhi);
#pragma unroll
    for (int j = 0; j < 4; ++j) {
      v16bf b = load_b_frag(curB, 32, wn * 64 + j * 16, 0, lcol, lhi);
      c[0][j] = wmma_bf16(a0, b, c[0][j]);
      c[1][j] = wmma_bf16(a1, b, c[1][j]);
    }
  }
#else
  (void)sA; (void)sB;
  for (int k0 = 0; k0 < D; k0 += 32) {
    v16bf a0 = load_a_frag(X, D, rowBase,      k0, lcol, lhi);
    v16bf a1 = load_a_frag(X, D, rowBase + 16, k0, lcol, lhi);
#pragma unroll
    for (int j = 0; j < 4; ++j) {
      v16bf b = load_b_frag(W, D, colBase + j * 16, k0, lcol, lhi);
      c[0][j] = wmma_bf16(a0, b, c[0][j]);
      c[1][j] = wmma_bf16(a1, b, c[1][j]);
    }
  }
#endif
#pragma unroll
  for (int i = 0; i < 2; ++i)
#pragma unroll
    for (int j = 0; j < 4; ++j)
#pragma unroll
      for (int e = 0; e < 8; ++e) {
        int r = rowBase + i * 16 + e + 8 * lhi;
        int o = colBase + j * 16 + lcol;
        float v = (c[i][j][e] + bias[o]) * scale;
        int b_ = r >> 12, n_ = r & (N - 1);
        int h_ = o >> 6, e_ = o & 63;
        out[(size_t)((b_ * H + h_) * N + n_) * DH + e_] = (bf16)v;
      }
}

// diag = 0.5 * sum(x^2) over head dim (inputs already scaled by NORM for q,k)
__global__ void k_rowsumsq(const bf16* __restrict__ Q, const bf16* __restrict__ K,
                           float* __restrict__ dQ, float* __restrict__ dK) {
  using namespace cfg;
  int i = blockIdx.x * 256 + threadIdx.x;
  int total = BH * N;
  if (i >= 2 * total) return;
  const bf16* src = (i < total) ? Q : K;
  float* dst = (i < total) ? dQ : dK;
  int r = (i < total) ? i : i - total;
  const bf16* p = src + (size_t)r * DH;
  float s = 0.f;
  for (int e = 0; e < DH; ++e) { float x = (float)p[e]; s += x * x; }
  dst[r] = 0.5f * s;
}

// feature GEMM core: rows nbase..+15, wave covers m-cols [wave*64, wave*64+64)
DEVI void feat_gemm(const bf16* __restrict__ Qh, const bf16* __restrict__ P,
                    int nbase, int wave, int lcol, int lhi, v8f c[4]) {
  using namespace cfg;
#pragma unroll
  for (int k0 = 0; k0 < DH; k0 += 32) {
    v16bf a = load_a_frag(Qh, DH, nbase, k0, lcol, lhi);
#pragma unroll
    for (int j = 0; j < 4; ++j) {
      v16bf b = load_b_frag(P, DH, wave * 64 + j * 16, k0, lcol, lhi);
      c[j] = wmma_bf16(a, b, c[j]);
    }
  }
}

// pass 1 for k: compute dash tiles, reduce block max only
__global__ __launch_bounds__(256) void k_kdash_max(const bf16* __restrict__ Kbf,
                                                   const bf16* __restrict__ Pk,
                                                   float* __restrict__ partmax) {
  using namespace cfg;
  int tid = threadIdx.x, wave = tid >> 5, lane = tid & 31;
  int lcol = lane & 15, lhi = lane >> 4;
  int bh = blockIdx.x >> 8;
  int nbase = (blockIdx.x & 255) * 16;
  const bf16* Qh = Kbf + (size_t)bh * N * DH;
  v8f c[4];
#pragma unroll
  for (int j = 0; j < 4; ++j) c[j] = vzero8();
  feat_gemm(Qh, Pk, nbase, wave, lcol, lhi, c);
  float m = -3.0e38f;
#pragma unroll
  for (int j = 0; j < 4; ++j)
#pragma unroll
    for (int e = 0; e < 8; ++e) m = fmaxf(m, c[j][e]);
  m = wave_max(m);
  __shared__ float sm[8];
  if (lane == 0) sm[wave] = m;
  __syncthreads();
  if (tid == 0) {
    float mm = sm[0];
    for (int w = 1; w < 8; ++w) mm = fmaxf(mm, sm[w]);
    partmax[blockIdx.x] = mm;
  }
}

__global__ void k_reduce_max(const float* __restrict__ pm, float* __restrict__ stab, int n) {
  float m = -3.0e38f;
  for (int i = threadIdx.x; i < n; i += 256) m = fmaxf(m, pm[i]);
  m = wave_max(m);
  __shared__ float sm[8];
  if ((threadIdx.x & 31) == 0) sm[threadIdx.x >> 5] = m;
  __syncthreads();
  if (threadIdx.x == 0) {
    float mm = sm[0];
    for (int w = 1; w < 8; ++w) mm = fmaxf(mm, sm[w]);
    stab[0] = mm;
  }
}

// q features: per-row max stabilizer (block owns 16 rows x all 512 features)
__global__ __launch_bounds__(256) void k_feat_q(const bf16* __restrict__ Qbf,
                                                const bf16* __restrict__ Pq,
                                                const float* __restrict__ diag,
                                                bf16* __restrict__ qp) {
  using namespace cfg;
  int tid = threadIdx.x, wave = tid >> 5, lane = tid & 31;
  int lcol = lane & 15, lhi = lane >> 4;
  int bh = blockIdx.x >> 8;
  int nbase = (blockIdx.x & 255) * 16;
  const bf16* Qh = Qbf + (size_t)bh * N * DH;
  v8f c[4];
#pragma unroll
  for (int j = 0; j < 4; ++j) c[j] = vzero8();
  feat_gemm(Qh, Pq, nbase, wave, lcol, lhi, c);

  float rmax[8];
#pragma unroll
  for (int e = 0; e < 8; ++e)
    rmax[e] = fmaxf(fmaxf(c[0][e], c[1][e]), fmaxf(c[2][e], c[3][e]));
#pragma unroll
  for (int mask = 1; mask < 16; mask <<= 1)
#pragma unroll
    for (int e = 0; e < 8; ++e)
      rmax[e] = fmaxf(rmax[e], __shfl_xor(rmax[e], mask, 32));

  __shared__ float smax[8][16];
  __shared__ float fmaxr[16];
  if (lcol == 0) {
#pragma unroll
    for (int e = 0; e < 8; ++e) smax[wave][e + 8 * lhi] = rmax[e];
  }
  __syncthreads();
  if (tid < 16) {
    float mm = smax[0][tid];
    for (int w = 1; w < 8; ++w) mm = fmaxf(mm, smax[w][tid]);
    fmaxr[tid] = mm;
  }
  __syncthreads();

  size_t rowbase = (size_t)bh * N + nbase;
#pragma unroll
  for (int e = 0; e < 8; ++e) {
    int r = e + 8 * lhi;
    float dg = diag[rowbase + r];
    float st = fmaxr[r];
#pragma unroll
    for (int j = 0; j < 4; ++j) {
      float v = RATIO * (__expf(c[j][e] - dg - st) + KEPS);
      qp[(rowbase + r) * M + wave * 64 + j * 16 + lcol] = (bf16)v;
    }
  }
}

// k features: global scalar stabilizer
__global__ __launch_bounds__(256) void k_feat_k(const bf16* __restrict__ Kbf,
                                                const bf16* __restrict__ Pk,
                                                const float* __restrict__ diag,
                                                const float* __restrict__ stab,
                                                bf16* __restrict__ kp) {
  using namespace cfg;
  int tid = threadIdx.x, wave = tid >> 5, lane = tid & 31;
  int lcol = lane & 15, lhi = lane >> 4;
  int bh = blockIdx.x >> 8;
  int nbase = (blockIdx.x & 255) * 16;
  const bf16* Qh = Kbf + (size_t)bh * N * DH;
  v8f c[4];
#pragma unroll
  for (int j = 0; j < 4; ++j) c[j] = vzero8();
  feat_gemm(Qh, Pk, nbase, wave, lcol, lhi, c);
  float st = stab[0];
  size_t rowbase = (size_t)bh * N + nbase;
#pragma unroll
  for (int e = 0; e < 8; ++e) {
    int r = e + 8 * lhi;
    float dg = diag[rowbase + r];
#pragma unroll
    for (int j = 0; j < 4; ++j) {
      float v = RATIO * (__expf(c[j][e] - dg - st) + KEPS);
      kp[(rowbase + r) * M + wave * 64 + j * 16 + lcol] = (bf16)v;
    }
  }
}

__global__ void k_kcumsum(const bf16* __restrict__ kp, float* __restrict__ kcum) {
  using namespace cfg;
  int bh = blockIdx.x >> 1;
  int m = (blockIdx.x & 1) * 256 + threadIdx.x;
  const bf16* p = kp + (size_t)bh * N * M + m;
  float s = 0.f;
  for (int n = 0; n < N; ++n) s += (float)p[(size_t)n * M];
  kcum[bh * M + m] = s;
}

__global__ __launch_bounds__(256) void k_dinv(const bf16* __restrict__ qp,
                                              const float* __restrict__ kcum,
                                              float* __restrict__ dinv) {
  using namespace cfg;
  int wave = threadIdx.x >> 5, lane = threadIdx.x & 31;
  size_t rg = (size_t)blockIdx.x * 8 + wave;
  int bh = (int)(rg >> 12);
  const bf16* p = qp + rg * M;
  const float* kc = kcum + bh * M;
  float s = 0.f;
  for (int m = lane; m < M; m += 32) s += (float)p[m] * kc[m];
  s = wave_sum(s);
  if (lane == 0) dinv[rg] = 1.f / s;
}

// context^T[e,m] = sum_n k'[n,m] * v[n,e] -- both operands transposed via LDS
__global__ __launch_bounds__(256) void k_context(const bf16* __restrict__ kp,
                                                 const bf16* __restrict__ Vbf,
                                                 bf16* __restrict__ ctxT) {
  using namespace cfg;
  __shared__ bf16 sK[128 * 32];  // [m][n]
  __shared__ bf16 sV[64 * 32];   // [e][n]
  int tid = threadIdx.x, wave = tid >> 5, lane = tid & 31;
  int lcol = lane & 15, lhi = lane >> 4;
  int bh = blockIdx.y;
  int mb = blockIdx.x * 128;
  const bf16* kph = kp + (size_t)bh * N * M;
  const bf16* vh  = Vbf + (size_t)bh * N * DH;
  v8f c[4];
#pragma unroll
  for (int j = 0; j < 4; ++j) c[j] = vzero8();

  for (int n0 = 0; n0 < N; n0 += 32) {
    __syncthreads();
    {
      int nl = tid >> 3, ms = (tid & 7) * 16;
      v16bf t = *reinterpret_cast<const v16bf*>(kph + (size_t)(n0 + nl) * M + mb + ms);
#pragma unroll
      for (int j2 = 0; j2 < 16; ++j2) sK[(ms + j2) * 32 + nl] = t[j2];
    }
    if (tid < 128) {
      int nl = tid >> 2, es = (tid & 3) * 16;
      v16bf t = *reinterpret_cast<const v16bf*>(vh + (size_t)(n0 + nl) * DH + es);
#pragma unroll
      for (int j2 = 0; j2 < 16; ++j2) sV[(es + j2) * 32 + nl] = t[j2];
    }
    __syncthreads();
    v16bf a = load_a_frag(sK, 32, wave * 16, 0, lcol, lhi);
#pragma unroll
    for (int j = 0; j < 4; ++j) {
      v16bf b = load_b_frag(sV, 32, j * 16, 0, lcol, lhi);
      c[j] = wmma_bf16(a, b, c[j]);
    }
  }
#pragma unroll
  for (int j = 0; j < 4; ++j)
#pragma unroll
    for (int e = 0; e < 8; ++e) {
      int mrow = mb + wave * 16 + e + 8 * lhi;
      int ecol = j * 16 + lcol;
      ctxT[((size_t)bh * DH + ecol) * M + mrow] = (bf16)c[j][e];
    }
}

// out[n,e] = d_inv[n] * sum_m q'[n,m] * ctxT[e,m]; merge heads into [B,N,D]
__global__ __launch_bounds__(256) void k_attn_out(const bf16* __restrict__ qp,
                                                  const bf16* __restrict__ ctxT,
                                                  const float* __restrict__ dinv,
                                                  bf16* __restrict__ ctxm) {
  using namespace cfg;
  int tid = threadIdx.x, wave = tid >> 5, lane = tid & 31;
  int lcol = lane & 15, lhi = lane >> 4;
  int bh = blockIdx.y;
  int b_ = bh >> 3, h_ = bh & 7;
  int nb = blockIdx.x * 128 + wave * 16;
  const bf16* qph = qp + (size_t)bh * N * M;
  const bf16* cT  = ctxT + (size_t)bh * DH * M;
  v8f c[4];
#pragma unroll
  for (int j = 0; j < 4; ++j) c[j] = vzero8();
  for (int k0 = 0; k0 < M; k0 += 32) {
    __builtin_prefetch(qph + (size_t)(nb + lcol) * M + k0 + 64, 0, 1);
    v16bf a = load_a_frag(qph, M, nb, k0, lcol, lhi);
#pragma unroll
    for (int j = 0; j < 4; ++j) {
      v16bf b = load_b_frag(cT, M, j * 16, k0, lcol, lhi);
      c[j] = wmma_bf16(a, b, c[j]);
    }
  }
#pragma unroll
  for (int e = 0; e < 8; ++e) {
    int nloc = nb + e + 8 * lhi;
    float di = dinv[(size_t)bh * N + nloc];
#pragma unroll
    for (int j = 0; j < 4; ++j) {
      int ecol = j * 16 + lcol;
      ctxm[((size_t)b_ * N + nloc) * D + h_ * DH + ecol] = (bf16)(c[j][e] * di);
    }
  }
}

// output projection + bias + residual (fp32 out for LN)
__global__ __launch_bounds__(256) void k_gemm_out(const bf16* __restrict__ A,
                                                  const bf16* __restrict__ W,
                                                  const float* __restrict__ bias,
                                                  const float* __restrict__ resid,
                                                  float* __restrict__ xres) {
  using namespace cfg;
  __shared__ bf16 sA[2 * 128 * 32];
  __shared__ bf16 sB[2 * 128 * 32];
  const int tid = threadIdx.x, wave = tid >> 5, lane = tid & 31;
  const int lcol = lane & 15, lhi = lane >> 4;
  const int wm = wave & 3, wn = wave >> 2;
  const int rowBlock = blockIdx.x * 128, colBlock = blockIdx.y * 128;
  const int rowBase = rowBlock + wm * 32, colBase = colBlock + wn * 64;
  v8f c[2][4];
#pragma unroll
  for (int i = 0; i < 2; ++i)
#pragma unroll
    for (int j = 0; j < 4; ++j) c[i][j] = vzero8();

#if ASYNC_LDS
  issue_tile(A, W, D, sA, sB, rowBlock, colBlock, 0, tid);
  for (int ks = 0; ks < D / 32; ++ks) {
    wait_async0();
    __syncthreads();
    bf16* curA = sA + (ks & 1) * (128 * 32);
    bf16* curB = sB + (ks & 1) * (128 * 32);
    if (ks + 1 < D / 32)
      issue_tile(A, W, D, sA + ((ks + 1) & 1) * (128 * 32),
                 sB + ((ks + 1) & 1) * (128 * 32),
                 rowBlock, colBlock, (ks + 1) * 32, tid);
    v16bf a0 = load_a_frag(curA, 32, wm * 32,      0, lcol, lhi);
    v16bf a1 = load_a_frag(curA, 32, wm * 32 + 16, 0, lcol, lhi);
#pragma unroll
    for (int j = 0; j < 4; ++j) {
      v16bf b = load_b_frag(curB, 32, wn * 64 + j * 16, 0, lcol, lhi);
      c[0][j] = wmma_bf16(a0, b, c[0][j]);
      c[1][j] = wmma_bf16(a1, b, c[1][j]);
    }
  }
#else
  (void)sA; (void)sB;
  for (int k0 = 0; k0 < D; k0 += 32) {
    v16bf a0 = load_a_frag(A, D, rowBase,      k0, lcol, lhi);
    v16bf a1 = load_a_frag(A, D, rowBase + 16, k0, lcol, lhi);
#pragma unroll
    for (int j = 0; j < 4; ++j) {
      v16bf b = load_b_frag(W, D, colBase + j * 16, k0, lcol, lhi);
      c[0][j] = wmma_bf16(a0, b, c[0][j]);
      c[1][j] = wmma_bf16(a1, b, c[1][j]);
    }
  }
#endif
#pragma unroll
  for (int i = 0; i < 2; ++i)
#pragma unroll
    for (int j = 0; j < 4; ++j)
#pragma unroll
      for (int e = 0; e < 8; ++e) {
        int r = rowBase + i * 16 + e + 8 * lhi;
        int o = colBase + j * 16 + lcol;
        float v = c[i][j][e] + bias[o] + resid[(size_t)r * D + o];
        xres[(size_t)r * D + o] = v;
      }
}

__global__ __launch_bounds__(256) void k_layernorm(const float* __restrict__ xr,
                                                   const float* __restrict__ gamma,
                                                   const float* __restrict__ beta,
                                                   float* __restrict__ out) {
  using namespace cfg;
  int r = blockIdx.x, tid = threadIdx.x;
  const float* row = xr + (size_t)r * D;
  float x0 = row[tid], x1 = row[tid + 256];
  float s = wave_sum(x0 + x1);
  float ss = wave_sum(x0 * x0 + x1 * x1);
  __shared__ float s1[8], s2[8];
  if ((tid & 31) == 0) { s1[tid >> 5] = s; s2[tid >> 5] = ss; }
  __syncthreads();
  float ts = 0.f, tss = 0.f;
  for (int w = 0; w < 8; ++w) { ts += s1[w]; tss += s2[w]; }
  float mu = ts * (1.f / D);
  float var = tss * (1.f / D) - mu * mu;
  float inv = rsqrtf(var + LNEPS);
  out[(size_t)r * D + tid]       = (x0 - mu) * inv * gamma[tid] + beta[tid];
  out[(size_t)r * D + tid + 256] = (x1 - mu) * inv * gamma[tid + 256] + beta[tid + 256];
}

// ---------------------------------------------------------------------------
extern "C" void kernel_launch(void* const* d_in, const int* in_sizes, int n_in,
                              void* d_out, int out_size, void* d_ws, size_t ws_size,
                              hipStream_t stream) {
  using namespace cfg;
  (void)in_sizes; (void)n_in; (void)out_size; (void)ws_size;
  const float* x   = (const float*)d_in[0];
  const float* Wq  = (const float*)d_in[2];
  const float* bq  = (const float*)d_in[3];
  const float* Wk  = (const float*)d_in[4];
  const float* bk  = (const float*)d_in[5];
  const float* Wv  = (const float*)d_in[6];
  const float* bv  = (const float*)d_in[7];
  const float* pq  = (const float*)d_in[8];
  const float* pk  = (const float*)d_in[9];
  const float* Wd  = (const float*)d_in[10];
  const float* bd  = (const float*)d_in[11];
  const float* gm  = (const float*)d_in[12];
  const float* bt  = (const float*)d_in[13];
  float* out = (float*)d_out;

  char* w = (char*)d_ws;
  auto alloc = [&](size_t bytes) -> char* {
    char* p = w; w += (bytes + 255) & ~(size_t)255; return p;
  };
  bf16* Xbf  = (bf16*)alloc((size_t)BN * D * 2);
  bf16* Wqb  = (bf16*)alloc((size_t)D * D * 2);
  bf16* Wkb  = (bf16*)alloc((size_t)D * D * 2);
  bf16* Wvb  = (bf16*)alloc((size_t)D * D * 2);
  bf16* Wdb  = (bf16*)alloc((size_t)D * D * 2);
  bf16* Pqb  = (bf16*)alloc((size_t)M * DH * 2);
  bf16* Pkb  = (bf16*)alloc((size_t)M * DH * 2);
  bf16* Qbf  = (bf16*)alloc((size_t)BH * N * DH * 2);
  bf16* Kbf  = (bf16*)alloc((size_t)BH * N * DH * 2);
  bf16* Vbf  = (bf16*)alloc((size_t)BH * N * DH * 2);
  float* diagQ = (float*)alloc((size_t)BH * N * 4);
  float* diagK = (float*)alloc((size_t)BH * N * 4);
  bf16* qp   = (bf16*)alloc((size_t)BH * N * M * 2);
  bf16* kp   = (bf16*)alloc((size_t)BH * N * M * 2);
  float* kcum = (float*)alloc((size_t)BH * M * 4);
  float* dinv = (float*)alloc((size_t)BH * N * 4);
  bf16* ctxT = (bf16*)alloc((size_t)BH * DH * M * 2);
  bf16* ctxm = (bf16*)alloc((size_t)BN * D * 2);
  float* xres = (float*)alloc((size_t)BN * D * 4);
  float* partmax = (float*)alloc((size_t)BH * (N / 16) * 4);
  float* stab = (float*)alloc(256);

  k_cvt_bf16<<<(BN * D + 255) / 256, 256, 0, stream>>>(x, Xbf, BN * D);
  k_cvt_bf16<<<(D * D + 255) / 256, 256, 0, stream>>>(Wq, Wqb, D * D);
  k_cvt_bf16<<<(D * D + 255) / 256, 256, 0, stream>>>(Wk, Wkb, D * D);
  k_cvt_bf16<<<(D * D + 255) / 256, 256, 0, stream>>>(Wv, Wvb, D * D);
  k_cvt_bf16<<<(D * D + 255) / 256, 256, 0, stream>>>(Wd, Wdb, D * D);
  k_cvt_bf16<<<(M * DH + 255) / 256, 256, 0, stream>>>(pq, Pqb, M * DH);
  k_cvt_bf16<<<(M * DH + 255) / 256, 256, 0, stream>>>(pk, Pkb, M * DH);

  dim3 g1(BN / 128, D / 128);
  k_gemm_qkv<<<g1, 256, 0, stream>>>(Xbf, Wqb, bq, Qbf, NORM);
  k_gemm_qkv<<<g1, 256, 0, stream>>>(Xbf, Wkb, bk, Kbf, NORM);
  k_gemm_qkv<<<g1, 256, 0, stream>>>(Xbf, Wvb, bv, Vbf, 1.0f);

  k_rowsumsq<<<(2 * BH * N) / 256, 256, 0, stream>>>(Qbf, Kbf, diagQ, diagK);

  k_kdash_max<<<BH * (N / 16), 256, 0, stream>>>(Kbf, Pkb, partmax);
  k_reduce_max<<<1, 256, 0, stream>>>(partmax, stab, BH * (N / 16));

  k_feat_q<<<BH * (N / 16), 256, 0, stream>>>(Qbf, Pqb, diagQ, qp);
  k_feat_k<<<BH * (N / 16), 256, 0, stream>>>(Kbf, Pkb, diagK, stab, kp);

  k_kcumsum<<<BH * (M / 256), 256, 0, stream>>>(kp, kcum);
  k_dinv<<<(BH * N) / 8, 256, 0, stream>>>(qp, kcum, dinv);

  k_context<<<dim3(M / 128, BH), 256, 0, stream>>>(kp, Vbf, ctxT);
  k_attn_out<<<dim3(N / 128, BH), 256, 0, stream>>>(qp, ctxT, dinv, ctxm);

  k_gemm_out<<<g1, 256, 0, stream>>>(ctxm, Wdb, bd, x, xres);
  k_layernorm<<<BN, 256, 0, stream>>>(xres, gm, bt, out);
}
